// Network_1400159338849
// MI455X (gfx1250) — matrix operand order repaired and checked by reading, not compiled, not run on hardware
//
#include <hip/hip_runtime.h>
#include <hip/hip_bf16.h>

typedef __attribute__((ext_vector_type(16))) _Float16 hvec16;
typedef __attribute__((ext_vector_type(8)))  _Float16 hvec8;
typedef __attribute__((ext_vector_type(8)))  float    fvec8;

#define B_     64
#define N_IN   2312
#define N_HID  512
#define N_OUT  10
#define T_     300
#define TP     304          // T padded to 16
#define IP     2336         // N_IN padded to 32
#define TTILES (TP / 16)    // 19
#define OTILES (N_HID / 16) // 32

// SLAYER constants (theta=10, tauSr=10, tauRef=1, scaleRef=2, Ts=1)
#define THETA   10.0f
#define A_SR    0.904837418f   // exp(-1/10)
#define A_REF   0.367879441f   // exp(-1)
#define C_SR    0.271828183f   // e / tauSr
#define C_REF  -54.3656366f    // -scaleRef*theta*e

// ---------------------------------------------------------------------------
// 1) transpose + f32->f16: x[b, i, t] -> xt[b, t, i] (padded, zero-filled)
// ---------------------------------------------------------------------------
__global__ __launch_bounds__(256) void xconv_kernel(const float* __restrict__ x,
                                                    _Float16* __restrict__ xt) {
  __shared__ _Float16 tile[32][33];
  const int b  = blockIdx.z;
  const int i0 = blockIdx.x * 32;
  const int t0 = blockIdx.y * 32;
  const int tx = threadIdx.x, ty = threadIdx.y;   // block (32, 8)
  #pragma unroll
  for (int r = 0; r < 4; ++r) {
    int i = i0 + ty + r * 8;
    int t = t0 + tx;
    float v = (i < N_IN && t < T_) ? x[((size_t)b * N_IN + i) * T_ + t] : 0.0f;
    tile[ty + r * 8][tx] = (_Float16)v;
  }
  __syncthreads();
  #pragma unroll
  for (int r = 0; r < 4; ++r) {
    int t = t0 + ty + r * 8;
    int i = i0 + tx;
    if (t < TP && i < IP)
      xt[((size_t)b * TP + t) * IP + i] = tile[tx][ty + r * 8];
  }
}

// ---------------------------------------------------------------------------
// 2) weight converts (with K / row padding)
// ---------------------------------------------------------------------------
__global__ __launch_bounds__(256) void w1conv_kernel(const float* __restrict__ w1,
                                                     _Float16* __restrict__ w1h) {
  int id = blockIdx.x * blockDim.x + threadIdx.x;
  if (id >= N_HID * IP) return;
  int o = id / IP, i = id % IP;
  w1h[id] = (i < N_IN) ? (_Float16)w1[(size_t)o * N_IN + i] : (_Float16)0.0f;
}

__global__ __launch_bounds__(256) void w2conv_kernel(const float* __restrict__ w2,
                                                     _Float16* __restrict__ w2h) {
  int id = blockIdx.x * blockDim.x + threadIdx.x;
  if (id >= 16 * N_HID) return;
  int j = id / N_HID, o = id % N_HID;
  w2h[id] = (j < N_OUT) ? (_Float16)w2[(size_t)j * N_HID + o] : (_Float16)0.0f;
}

// zero the t-padding rows of the spike buffer s1h[b, t(300..303), o]
__global__ __launch_bounds__(256) void s1pad_kernel(_Float16* __restrict__ s1h) {
  int id = blockIdx.x * blockDim.x + threadIdx.x;
  if (id >= B_ * 4 * N_HID) return;
  int b = id / (4 * N_HID);
  int r = id % (4 * N_HID);
  int t = T_ + r / N_HID;
  int o = r % N_HID;
  s1h[((size_t)b * TP + t) * N_HID + o] = (_Float16)0.0f;
}

// ---------------------------------------------------------------------------
// 3) GEMM1: a1[b, o, t] = sum_i xt[b, t, i] * w1h[o, i]  (WMMA f16 -> f32)
//    per-wave 16x16 tile: M = t, N = o, K-loop over IP in steps of 32
// ---------------------------------------------------------------------------
__global__ __launch_bounds__(256) void gemm1_kernel(const _Float16* __restrict__ xt,
                                                    const _Float16* __restrict__ w1h,
                                                    float* __restrict__ a1) {
  const int lane = threadIdx.x & 31;
  const int wave = threadIdx.x >> 5;                 // 8 waves / block
  const int b    = blockIdx.x / TTILES;
  const int t0   = (blockIdx.x % TTILES) * 16;
  const int o0   = (blockIdx.y * 8 + wave) * 16;
  const int g    = lane >> 4;                        // lane group (K-half select)
  const int ln   = lane & 15;                        // M (row) / N (col) index

  const _Float16* arow = xt  + ((size_t)b * TP + t0 + ln) * IP;  // A row = t0+ln
  const _Float16* brow = w1h + ((size_t)(o0 + ln)) * IP;         // B col = o0+ln

  fvec8 c = {};
  for (int k0 = 0; k0 < IP; k0 += 32) {
    hvec8 alo = *(const hvec8*)(arow + k0 + 8 * g);        // K 0..7  / 8..15
    hvec8 ahi = *(const hvec8*)(arow + k0 + 16 + 8 * g);   // K 16..23 / 24..31
    hvec16 amat = __builtin_shufflevector(alo, ahi, 0, 1, 2, 3, 4, 5, 6, 7,
                                          8, 9, 10, 11, 12, 13, 14, 15);
    hvec8 blo = *(const hvec8*)(brow + k0 + 8 * g);        // K 0..7  / 8..15
    hvec8 bhi = *(const hvec8*)(brow + k0 + 16 + 8 * g);   // K 16..23 / 24..31
    hvec16 bmat = __builtin_shufflevector(blo, bhi, 0, 1, 2, 3, 4, 5, 6, 7,
                                          8, 9, 10, 11, 12, 13, 14, 15);
    c = __builtin_amdgcn_wmma_f32_16x16x32_f16(false, amat, false, bmat,
                                               (short)0, c, false, false);
  }
  // D layout: lane ln -> N = o0+ln ; rows M = t0 + 8*g + v (v = VGPR 0..7)
  const int o  = o0 + ln;
  const int tb = t0 + 8 * g;
  float* out = a1 + ((size_t)b * N_HID + o) * T_ + tb;
  if (tb + 8 <= T_) {
    *(float4*)(out)     = make_float4(c[0], c[1], c[2], c[3]);
    *(float4*)(out + 4) = make_float4(c[4], c[5], c[6], c[7]);
  } else {
    #pragma unroll
    for (int v = 0; v < 8; ++v)
      if (tb + v < T_) out[v] = c[v];
  }
}

// ---------------------------------------------------------------------------
// 4) neuron scan: psp (alpha IIR) + refractory (alpha IIR) + threshold
//    a1[b, o, 0..T) -> spikes s1h[b, t, o] (f16)
// ---------------------------------------------------------------------------
__global__ __launch_bounds__(256) void neuron1_kernel(const float* __restrict__ a1,
                                                      _Float16* __restrict__ s1h) {
  int id = blockIdx.x * blockDim.x + threadIdx.x;
  if (id >= B_ * N_HID) return;
  const int b = id / N_HID, o = id % N_HID;
  const float* u = a1 + (size_t)id * T_;
  _Float16* sp = s1h + (size_t)b * TP * N_HID + o;
  float p0 = 0.f, p1 = 0.f;   // psp alpha-filter states
  float q0 = 0.f, q1 = 0.f;   // refractory alpha-filter states
  for (int t = 0; t < T_; ++t) {
    p1 = A_SR * (p1 + p0);           // s1[t] = a*(s1[t-1]+s0[t-1])
    p0 = A_SR * p0 + u[t];           // s0[t] = a*s0[t-1] + x[t]
    q1 = A_REF * (q1 + q0);
    q0 = A_REF * q0;
    float upost = C_SR * p1 + C_REF * q1;
    float s = (upost >= THETA) ? 1.0f : 0.0f;   // /Ts with Ts=1
    q0 += s;                          // spike enters refractory state
    sp[(size_t)t * N_HID] = (_Float16)s;
  }
}

// ---------------------------------------------------------------------------
// 5) GEMM2: a2[b, j, t] = sum_o s1h[b, t, o] * w2h[j, o]  (WMMA, N padded 16)
// ---------------------------------------------------------------------------
__global__ __launch_bounds__(32) void gemm2_kernel(const _Float16* __restrict__ s1h,
                                                   const _Float16* __restrict__ w2h,
                                                   float* __restrict__ a2) {
  const int lane = threadIdx.x & 31;
  const int b    = blockIdx.x / TTILES;
  const int t0   = (blockIdx.x % TTILES) * 16;
  const int g    = lane >> 4;
  const int ln   = lane & 15;

  const _Float16* arow = s1h + ((size_t)b * TP + t0 + ln) * N_HID;
  const _Float16* brow = w2h + (size_t)ln * N_HID;

  fvec8 c = {};
  for (int k0 = 0; k0 < N_HID; k0 += 32) {
    hvec8 alo = *(const hvec8*)(arow + k0 + 8 * g);
    hvec8 ahi = *(const hvec8*)(arow + k0 + 16 + 8 * g);
    hvec16 amat = __builtin_shufflevector(alo, ahi, 0, 1, 2, 3, 4, 5, 6, 7,
                                          8, 9, 10, 11, 12, 13, 14, 15);
    hvec8 blo = *(const hvec8*)(brow + k0 + 8 * g);
    hvec8 bhi = *(const hvec8*)(brow + k0 + 16 + 8 * g);
    hvec16 bmat = __builtin_shufflevector(blo, bhi, 0, 1, 2, 3, 4, 5, 6, 7,
                                          8, 9, 10, 11, 12, 13, 14, 15);
    c = __builtin_amdgcn_wmma_f32_16x16x32_f16(false, amat, false, bmat,
                                               (short)0, c, false, false);
  }
  const int j  = ln;
  const int tb = t0 + 8 * g;
  if (j < N_OUT) {
    float* out = a2 + ((size_t)b * N_OUT + j) * T_ + tb;
    #pragma unroll
    for (int v = 0; v < 8; ++v)
      if (tb + v < T_) out[v] = c[v];
  }
}

// ---------------------------------------------------------------------------
// 6) output neuron scan -> d_out[b, j, t] (f32)
// ---------------------------------------------------------------------------
__global__ __launch_bounds__(64) void neuron2_kernel(const float* __restrict__ a2,
                                                     float* __restrict__ out) {
  int id = blockIdx.x * blockDim.x + threadIdx.x;
  if (id >= B_ * N_OUT) return;
  const float* u = a2 + (size_t)id * T_;
  float* sp = out + (size_t)id * T_;
  float p0 = 0.f, p1 = 0.f, q0 = 0.f, q1 = 0.f;
  for (int t = 0; t < T_; ++t) {
    p1 = A_SR * (p1 + p0);
    p0 = A_SR * p0 + u[t];
    q1 = A_REF * (q1 + q0);
    q0 = A_REF * q0;
    float upost = C_SR * p1 + C_REF * q1;
    float s = (upost >= THETA) ? 1.0f : 0.0f;
    q0 += s;
    sp[t] = s;
  }
}

// ---------------------------------------------------------------------------
extern "C" void kernel_launch(void* const* d_in, const int* in_sizes, int n_in,
                              void* d_out, int out_size, void* d_ws, size_t ws_size,
                              hipStream_t stream) {
  const float* x  = (const float*)d_in[0];   // [64, 2312, 300]
  const float* w1 = (const float*)d_in[1];   // [512, 2312]
  const float* w2 = (const float*)d_in[2];   // [10, 512]
  float* out = (float*)d_out;                // [64, 10, 300]

  char* ws = (char*)d_ws;
  // layout (bytes); s1h aliases xt (xt is dead once gemm1 completes)
  const size_t XT_BYTES  = (size_t)B_ * TP * IP * 2;        // 90,898,432
  const size_t W1H_OFF   = XT_BYTES;
  const size_t W1H_BYTES = (size_t)N_HID * IP * 2;          //  2,392,064
  const size_t W2H_OFF   = W1H_OFF + W1H_BYTES;
  const size_t W2H_BYTES = (size_t)16 * N_HID * 2;          //     16,384
  const size_t A1_OFF    = W2H_OFF + W2H_BYTES;
  const size_t A1_BYTES  = (size_t)B_ * N_HID * T_ * 4;     // 39,321,600
  const size_t A2_OFF    = A1_OFF + A1_BYTES;

  _Float16* xt  = (_Float16*)(ws);
  _Float16* s1h = (_Float16*)(ws);           // alias over xt
  _Float16* w1h = (_Float16*)(ws + W1H_OFF);
  _Float16* w2h = (_Float16*)(ws + W2H_OFF);
  float*    a1  = (float*)(ws + A1_OFF);
  float*    a2  = (float*)(ws + A2_OFF);

  // 1) transpose/convert input spikes
  {
    dim3 grid(IP / 32, (TP + 31) / 32, B_);  // (73, 10, 64)
    dim3 block(32, 8);
    xconv_kernel<<<grid, block, 0, stream>>>(x, xt);
  }
  // 2) weight converts
  w1conv_kernel<<<(N_HID * IP + 255) / 256, 256, 0, stream>>>(w1, w1h);
  w2conv_kernel<<<(16 * N_HID + 255) / 256, 256, 0, stream>>>(w2, w2h);
  // 3) GEMM1 (WMMA)
  {
    dim3 grid(B_ * TTILES, OTILES / 8);      // (1216, 4), 8 waves/block
    gemm1_kernel<<<grid, 256, 0, stream>>>(xt, w1h, a1);
  }
  // 4) hidden-layer neuron scan (writes s1h over the xt alias)
  neuron1_kernel<<<(B_ * N_HID + 255) / 256, 256, 0, stream>>>(a1, s1h);
  // zero the 4 padded t-rows of s1h (read by gemm2)
  s1pad_kernel<<<(B_ * 4 * N_HID + 255) / 256, 256, 0, stream>>>(s1h);
  // 5) GEMM2 (WMMA)
  gemm2_kernel<<<B_ * TTILES, 32, 0, stream>>>(s1h, w2h, a2);
  // 6) output neuron scan
  neuron2_kernel<<<(B_ * N_OUT + 63) / 64, 64, 0, stream>>>(a2, out);
}
